// RNN_78288663872468
// MI455X (gfx1250) — matrix operand order repaired and checked by reading, not compiled, not run on hardware
//
#include <hip/hip_runtime.h>
#include <hip/hip_bf16.h>
#include <math.h>

// Problem constants
#define T_  512
#define B_  32
#define I_  1024
#define H_  1024
#define G_  4096          // 4*H
#define KT_ 32            // K tiles of 32 (K = 1024)
#define REC_BLOCKS 64     // recurrent persistent blocks; each owns 16 h-columns

// Recurrent-kernel LDS layout (dynamic):
//   [0,       131072)  W_hh slice, fragment-blocked bf16 (4 nt * 32 kt * 512)
//   [131072,  139264)  gate exchange buffer [4][32][16] f32
//   [139264,  205824)  h staging: 32 rows, 2048B data + 16B pad per 1024B
#define GATES_OFF   131072
#define HSTAGE_OFF  139264
#define HROW_STRIDE 2080
#define REC_LDS     205824

typedef __attribute__((ext_vector_type(16))) __bf16 v16bf;
typedef __attribute__((ext_vector_type(8)))  __bf16 v8bf;
typedef __attribute__((ext_vector_type(8)))  float  v8f;
typedef __attribute__((ext_vector_type(4)))  unsigned int v4u;
typedef __attribute__((ext_vector_type(8)))  int v8i;
typedef __attribute__((ext_vector_type(4)))  int v4i;

#if defined(__has_builtin)
# if __has_builtin(__builtin_amdgcn_tensor_load_to_lds) && __has_builtin(__builtin_amdgcn_s_wait_tensorcnt)
#  define HAVE_TDM 1
# endif
#endif
#ifndef HAVE_TDM
# define HAVE_TDM 0
#endif

// ---------------------------------------------------------------------------
// Kernel 0a: pack fp32 weights into WMMA-B-fragment-blocked bf16 layout.
// B matrix is K x N (N = gate rows). Tile = 32(K) x 16(N) = 512 bf16; lane l
// owns 16 contiguous bf16: n_in = l&15, k_in = (l>>4)*8 + s (s<8), 16 + (l>>4)*8 + (s-8) (s>=8).
// ---------------------------------------------------------------------------
__global__ __launch_bounds__(256) void lstm_pack_w(
    const float* __restrict__ w_ih, const float* __restrict__ w_hh,
    __bf16* __restrict__ wihP, __bf16* __restrict__ whhP)
{
  size_t idx = (size_t)blockIdx.x * 256 + threadIdx.x;
  const size_t NPK = (size_t)G_ * I_;
  const float* src; __bf16* dst; size_t p;
  if (idx < NPK)          { src = w_ih; dst = wihP; p = idx; }
  else if (idx < 2 * NPK) { src = w_hh; dst = whhP; p = idx - NPK; }
  else return;
  int s   = (int)(p & 15);
  int l   = (int)((p >> 4) & 31);
  int kt  = (int)((p >> 9) & 31);
  int nt  = (int)(p >> 14);
  int g   = nt * 16 + (l & 15);
  int k   = kt * 32 + ((s < 8) ? ((l >> 4) * 8 + s) : (16 + (l >> 4) * 8 + (s - 8)));
  dst[p] = (__bf16)src[(size_t)g * 1024 + k];
}

// ---------------------------------------------------------------------------
// Kernel 0b: pack fp32 x into WMMA-A-fragment-blocked bf16 layout.
// A tile = 16(M) x 32(K) = 512 bf16; lane l = m + 16*hi owns slots
//   j<8 : k_in = hi*8 + j ;  j>=8 : k_in = 16 + hi*8 + (j-8)
// ---------------------------------------------------------------------------
__global__ __launch_bounds__(256) void lstm_pack_x(
    const float* __restrict__ x, __bf16* __restrict__ xP)
{
  size_t p = (size_t)blockIdx.x * 256 + threadIdx.x;   // 16M elements
  if (p >= (size_t)T_ * B_ * I_) return;
  int s   = (int)(p & 15);
  int l   = (int)((p >> 4) & 31);
  int kt  = (int)((p >> 9) & 31);
  int mt  = (int)(p >> 14);
  int m   = l & 15;
  int hi  = l >> 4;
  int kin = (s < 8) ? (hi * 8 + s) : (16 + hi * 8 + (s - 8));
  int row = mt * 16 + m;
  int k   = kt * 32 + kin;
  xP[p] = (__bf16)x[(size_t)row * I_ + k];
}

// ---------------------------------------------------------------------------
// Kernel 0c: init h double-buffer from h0 (bf16), reset barrier + scalar outs.
// ---------------------------------------------------------------------------
__global__ __launch_bounds__(256) void lstm_init(
    const float* __restrict__ h0, __bf16* __restrict__ hbuf,
    unsigned* __restrict__ bar, float* __restrict__ outS)
{
  int e = blockIdx.x * 256 + threadIdx.x;
  if (e < B_ * H_) {
    hbuf[e]           = (__bf16)h0[e];
    hbuf[B_ * H_ + e] = (__bf16)0.0f;
  }
  if (e == 0) {
    *bar = 0u;
    outS[0] = 0.0f; outS[1] = 0.0f; outS[2] = 0.0f; outS[3] = 0.0f;
  }
}

// ---------------------------------------------------------------------------
// Kernel 1: pre-activations for ALL timesteps:
//   pre[(t*B+b), g] = sum_i x[t,b,i]*w_ih[g,i] + b_ih[g] + b_hh[g]   (bf16)
// M = 16384, N = 4096, K = 1024.  Wave computes a 32(M) x 64(N) strip:
// 2 A-frags + 4 B-frags -> 8 WMMAs per K step, all operands pre-packed bf16.
// ---------------------------------------------------------------------------
__global__ __launch_bounds__(256) void lstm_pre_gemm(
    const __bf16* __restrict__ xP, const __bf16* __restrict__ wihP,
    const float* __restrict__ b_ih, const float* __restrict__ b_hh,
    __bf16* __restrict__ pre)
{
  const int lane  = threadIdx.x & 31;
  const int wv    = threadIdx.x >> 5;
  const int gw    = blockIdx.x * 8 + wv;      // 0..32767
  const int mpair = gw >> 6;                  // 0..511 -> mtiles 2*mpair, +1
  const int nquad = gw & 63;                  // ntiles nquad*4 .. +3
  const int m     = lane & 15;
  const int hi    = lane >> 4;
  const int mt0   = mpair * 2;

  v8f acc[2][4] = {};
  for (int kt = 0; kt < KT_; ++kt) {
    v16bf a0 = *(const v16bf*)(xP + (size_t)((mt0 + 0) * KT_ + kt) * 512 + lane * 16);
    v16bf a1 = *(const v16bf*)(xP + (size_t)((mt0 + 1) * KT_ + kt) * 512 + lane * 16);
#pragma unroll
    for (int f = 0; f < 4; ++f) {
      const int nt = nquad * 4 + f;
      v16bf b = *(const v16bf*)(wihP + (size_t)(nt * KT_ + kt) * 512 + lane * 16);
      acc[0][f] = __builtin_amdgcn_wmma_f32_16x16x32_bf16(
          false, a0, false, b, (short)0, acc[0][f], false, false);
      acc[1][f] = __builtin_amdgcn_wmma_f32_16x16x32_bf16(
          false, a1, false, b, (short)0, acc[1][f], false, false);
    }
  }
#pragma unroll
  for (int mi = 0; mi < 2; ++mi) {
#pragma unroll
    for (int f = 0; f < 4; ++f) {
      const int nt  = nquad * 4 + f;
      const int col = nt * 16 + m;
      const float bias = b_ih[col] + b_hh[col];
#pragma unroll
      for (int v = 0; v < 8; ++v) {
        const int r = (mt0 + mi) * 16 + v + hi * 8;
        pre[(size_t)r * G_ + col] = (__bf16)(acc[mi][f][v] + bias);
      }
    }
  }
}

// ---------------------------------------------------------------------------
// Kernel 2: persistent recurrent kernel. 64 blocks x 256 threads (8 waves).
// Block bk owns h columns [bk*16, bk*16+16). W_hh slice lives in LDS for all
// 512 steps; h_prev is staged into LDS each step via the Tensor Data Mover
// (bank-conflict-padded rows); cell state c lives in registers.
// Wave w: gate g = w>>1, mtile = w&1 -> one 16x16 C tile per wave per step.
// ---------------------------------------------------------------------------
__global__ __launch_bounds__(256) void lstm_recurrent(
    const __bf16* __restrict__ whhP, const __bf16* __restrict__ pre,
    const float* __restrict__ c0, __bf16* __restrict__ hbuf,
    unsigned* __restrict__ bar, float* __restrict__ out, int nblocks)
{
  extern __shared__ char smem[];
  __bf16* whh_lds = (__bf16*)smem;
  float*  gatesL  = (float*)(smem + GATES_OFF);

  const int tid   = threadIdx.x;
  const int lane  = tid & 31;
  const int wv    = tid >> 5;
  const int bk    = blockIdx.x;
  const int mtile = wv & 1;
  const int g     = wv >> 1;
  const int m     = lane & 15;
  const int hi    = lane >> 4;

  // --- stage this block's W_hh slice into LDS (128 KB) ---
  {
    const uint4* src = (const uint4*)whhP;
    uint4* dst = (uint4*)smem;
    for (int i = tid; i < 8192; i += 256) {
      int lnt = i >> 11;
      int rem = i & 2047;
      int gnt = lnt * 64 + bk;
      dst[(size_t)lnt * 2048 + rem] = src[(size_t)gnt * 2048 + rem];
    }
  }
  __syncthreads();

#if HAVE_TDM
  unsigned hstage_lds_addr;
  {
    typedef __attribute__((address_space(3))) char lds_char_t;
    hstage_lds_addr = (unsigned)(unsigned long long)(lds_char_t*)(smem + HSTAGE_OFF);
  }
#endif

  // --- per-thread cell state: 2 (b, n) pairs over the 32x16 owned slice ---
  float cst[2], hlast[2];
  int eb[2], en[2];
#pragma unroll
  for (int p = 0; p < 2; ++p) {
    int e = tid + p * 256;
    eb[p] = e >> 4; en[p] = e & 15;
    cst[p]   = c0[eb[p] * H_ + bk * 16 + en[p]];
    hlast[p] = 0.0f;
  }

  float gn = 0.0f, hn = 0.0f;
  int gnc = 0, hnc = 0;
  const int col = g * H_ + bk * 16 + m;

  for (int t = 0; t < T_; ++t) {
    const __bf16* hsrc = hbuf + (size_t)(t & 1) * (B_ * H_);
    __bf16*       hdst = hbuf + (size_t)((t + 1) & 1) * (B_ * H_);

#if HAVE_TDM
    // Async: TDM moves the 32x1024 bf16 h tile into LDS, inserting a 16B pad
    // after every 1024B so LDS rows stride 2080B (bank-conflict avoidance).
    if (wv == 0) {
      unsigned long long ga = (unsigned long long)(uintptr_t)hsrc;
      v4u g0 = { 1u,                                   // count=1
                 hstage_lds_addr,                      // lds_addr
                 (unsigned)(ga & 0xffffffffu),
                 (unsigned)(ga >> 32) | (2u << 30) };  // addr[56:32] | type=2
      v8i g1 = { (3 << 16) | (1 << 20) | (7 << 22) | (3 << 25),
                 //  data_size=8B | pad_en | pad_interval=1KB | pad_amount=16B
                 (256 << 16),   // tensor_dim0[15:0]=256 (8B units) in [63:48]
                 (32 << 16),    // tensor_dim0 hi=0, tensor_dim1=32 rows
                 (256 << 16),   // tensor_dim1 hi=0, tile_dim0=256
                 32,            // tile_dim1=32, tile_dim2=0
                 256,           // tensor_dim0_stride low32 (8B units)
                 0, 0 };
      v4i g2 = { 0, 0, 0, 0 };
      v4i g3 = { 0, 0, 0, 0 };
      v8i g4 = { 0, 0, 0, 0, 0, 0, 0, 0 };
      __builtin_amdgcn_tensor_load_to_lds(g0, g1, g2, g3, g4, 0);
    }
#else
    // Fallback: cooperative synchronous copy with the same padded layout.
    {
      const uint4* src = (const uint4*)hsrc;
      for (int i = tid; i < 4096; i += 256) {
        int row    = i >> 7;
        int within = (i & 127) * 16;
        int pad    = (within >= 1024) ? 16 : 0;
        *(uint4*)(smem + HSTAGE_OFF + row * HROW_STRIDE + within + pad) = src[i];
      }
    }
#endif

    // init accumulator from precomputed input-projection gates (overlaps TDM)
    v8f acc;
#pragma unroll
    for (int v = 0; v < 8; ++v) {
      int r = t * B_ + mtile * 16 + v + hi * 8;
      acc[v] = (float)pre[(size_t)r * G_ + col];
    }
    if (t + 1 < T_)
      __builtin_prefetch(pre + (size_t)((t + 1) * B_ + mtile * 16) * G_ + col, 0, 1);

#if HAVE_TDM
    if (wv == 0) __builtin_amdgcn_s_wait_tensorcnt(0);
#endif
    __syncthreads();   // h tile visible in LDS

    // recurrent GEMM: acc += h_prev (32xK) x Whh_slice (Kx16), A from LDS
    const char* hrowb = smem + HSTAGE_OFF + (size_t)(mtile * 16 + m) * HROW_STRIDE;
#pragma unroll 4
    for (int kt = 0; kt < KT_; ++kt) {
      const int k0   = kt * 32 + hi * 8;
      const int k1   = k0 + 16;
      const int off0 = 2 * k0 + ((k0 >= 512) ? 16 : 0);
      const int off1 = 2 * k1 + ((k1 >= 512) ? 16 : 0);
      v8bf lo = *(const v8bf*)(hrowb + off0);
      v8bf hp = *(const v8bf*)(hrowb + off1);
      v16bf a;
#pragma unroll
      for (int j = 0; j < 8; ++j) { a[j] = lo[j]; a[8 + j] = hp[j]; }
      v16bf b = *(const v16bf*)(whh_lds + (size_t)(g * KT_ + kt) * 512 + lane * 16);
      acc = __builtin_amdgcn_wmma_f32_16x16x32_bf16(
          false, a, false, b, (short)0, acc, false, false);
    }

    // exchange gates through LDS: [gate][b][n_local]
#pragma unroll
    for (int v = 0; v < 8; ++v) {
      int b = mtile * 16 + v + hi * 8;
      gatesL[(g * 32 + b) * 16 + m] = acc[v];
    }
    __syncthreads();

    // element-wise LSTM cell with thresholded output gate
#pragma unroll
    for (int p = 0; p < 2; ++p) {
      int b = eb[p], nl = en[p];
      float gi = gatesL[(0 * 32 + b) * 16 + nl];
      float gf = gatesL[(1 * 32 + b) * 16 + nl];
      float gg = gatesL[(2 * 32 + b) * 16 + nl];
      float go = gatesL[(3 * 32 + b) * 16 + nl];
      float ig = 1.0f / (1.0f + __expf(-gi));
      float fg = 1.0f / (1.0f + __expf(-gf));
      float cg = tanhf(gg);
      float og = 1.0f / (1.0f + __expf(-go));
      float ogn = (og > 0.1f) ? og : 0.0f;   // NORM_THRESH stats
      gn += ogn; gnc += (ogn != 0.0f) ? 1 : 0;
      float ogh = (og > 0.1f) ? og : 0.0f;   // HARD_THRESH replaces outgate
      hn += ogh; hnc += (ogh != 0.0f) ? 1 : 0;
      float cnew = fg * cst[p] + ig * cg;
      float hnew = ogh * tanhf(cnew);
      cst[p] = cnew; hlast[p] = hnew;
      int n = bk * 16 + nl;
      out[(size_t)t * (B_ * H_) + b * H_ + n] = hnew;
      hdst[b * H_ + n] = (__bf16)hnew;
    }
    __syncthreads();

    // --- device-wide step barrier ---
    if (tid == 0) {
      __threadfence();
      atomicAdd(bar, 1u);
      unsigned target = (unsigned)(t + 1) * (unsigned)nblocks;
      while (*(volatile unsigned*)bar < target) __builtin_amdgcn_s_sleep(1);
    }
    __syncthreads();
    __threadfence();
  }

  // --- finals: hT, cT, scalar reductions ---
  float* outHT = out + (size_t)T_ * B_ * H_;
  float* outCT = outHT + B_ * H_;
  float* outS  = outCT + B_ * H_;
#pragma unroll
  for (int p = 0; p < 2; ++p) {
    int idx = eb[p] * H_ + bk * 16 + en[p];
    outHT[idx] = hlast[p];
    outCT[idx] = cst[p];
  }
  __shared__ float sgn, shn;
  __shared__ int   sgnc, shnc;
  if (tid == 0) { sgn = 0.0f; shn = 0.0f; sgnc = 0; shnc = 0; }
  __syncthreads();
  atomicAdd(&sgn, gn);  atomicAdd(&shn, hn);
  atomicAdd(&sgnc, gnc); atomicAdd(&shnc, hnc);
  __syncthreads();
  if (tid == 0) {
    atomicAdd(&outS[0], sgn);
    atomicAdd((int*)&outS[1], sgnc);
    atomicAdd(&outS[2], shn);
    atomicAdd((int*)&outS[3], shnc);
  }
}

// ---------------------------------------------------------------------------
extern "C" void kernel_launch(void* const* d_in, const int* in_sizes, int n_in,
                              void* d_out, int out_size, void* d_ws, size_t ws_size,
                              hipStream_t stream)
{
  const float* x   = (const float*)d_in[0];
  const float* h0  = (const float*)d_in[1];
  const float* c0  = (const float*)d_in[2];
  const float* wih = (const float*)d_in[3];
  const float* whh = (const float*)d_in[4];
  const float* bih = (const float*)d_in[5];
  const float* bhh = (const float*)d_in[6];
  float* out = (float*)d_out;

  // workspace layout
  char* ws = (char*)d_ws;
  __bf16*   wihP = (__bf16*)(ws);                          //   8 MB
  __bf16*   whhP = (__bf16*)(ws + ((size_t)8   << 20));    //   8 MB
  __bf16*   xP   = (__bf16*)(ws + ((size_t)16  << 20));    //  32 MB
  __bf16*   pre  = (__bf16*)(ws + ((size_t)48  << 20));    // 128 MB
  __bf16*   hbuf = (__bf16*)(ws + ((size_t)176 << 20));    // 128 KB
  unsigned* bar  = (unsigned*)(ws + ((size_t)177 << 20));

  float* outS = out + (size_t)T_ * B_ * H_ + (size_t)2 * B_ * H_;
  (void)in_sizes; (void)n_in; (void)out_size; (void)ws_size;

  // 0) pack weights + x to bf16 fragment layouts; init h buffer / barrier
  lstm_pack_w<<<32768, 256, 0, stream>>>(wih, whh, wihP, whhP);
  lstm_pack_x<<<65536, 256, 0, stream>>>(x, xP);
  lstm_init<<<128, 256, 0, stream>>>(h0, hbuf, bar, outS);

  // 1) all-timestep input projection GEMM (bf16 WMMA, f32 accumulate)
  lstm_pre_gemm<<<4096, 256, 0, stream>>>(xP, wihP, bih, bhh, pre);

  // 2) persistent recurrent kernel: W_hh in LDS, h staged per step via TDM
  lstm_recurrent<<<REC_BLOCKS, 256, REC_LDS, stream>>>(
      whhP, pre, c0, hbuf, bar, out, REC_BLOCKS);
}
